// FeatureAdaptation_5720896438347
// MI455X (gfx1250) — compile-verified, matrix operand order checked by reading, not run on hardware
//
#include <hip/hip_runtime.h>
#include <hip/hip_bf16.h>

#define B_    2
#define CHI_  256
#define CHO_  256
#define H_    128
#define W_    128
#define K_    9
#define HW_   (H_ * W_)

typedef __attribute__((ext_vector_type(16))) __bf16 v16bf;
typedef __attribute__((ext_vector_type(8)))  float  v8f;

union Frag32B { uint4 q[2]; v16bf v; };

__device__ __forceinline__ unsigned short f2bf(float f) {
    unsigned int u = __float_as_uint(f);
    u += 0x7FFFu + ((u >> 16) & 1u);   // round-to-nearest-even
    return (unsigned short)(u >> 16);
}

// ---------------------------------------------------------------------------
// Pass 0: pack conv_w (CHO,CHI,3,3) f32 -> bf16 in exact WMMA A-fragment lane
// order. Chunk = tap*8 + cblk covers K-slice (tap, c in [cblk*32, cblk*32+32)).
// Layout: [chunk][ot(16)][lane(32)][run(2)][j(8)] bf16  (32 B per lane).
// A-matrix 16x32 bf16 lane layout: lane<16 -> M=lane, elems 0..7=K0..7,
// 8..15=K16..23 ; lane>=16 -> M=lane-16, elems 0..7=K8..15, 8..15=K24..31.
// ---------------------------------------------------------------------------
__global__ void fa_pack_weights(const float* __restrict__ conv_w,
                                unsigned short* __restrict__ wpack) {
    int idx = blockIdx.x * 256 + threadIdx.x;       // < 72*16*32*2*8 = 589824
    int j     = idx & 7;
    int run   = (idx >> 3) & 1;
    int lane  = (idx >> 4) & 31;
    int ot    = (idx >> 9) & 15;
    int chunk = idx >> 13;                           // 0..71
    int tap   = chunk >> 3;
    int cblk  = chunk & 7;
    int o  = ot * 16 + (lane & 15);
    int kk = run * 16 + ((lane >> 4) ? 8 : 0) + j;
    int c  = cblk * 32 + kk;
    float v = conv_w[((size_t)o * CHI_ + c) * K_ + tap];
    wpack[idx] = f2bf(v);
}

// ---------------------------------------------------------------------------
// Main fused kernel: deformable sampling + implicit GEMM via bf16 WMMA.
// Block: 256 threads (8 wave32). Tile: (b, h, 32 w-positions) x 256 out-ch.
// wave id wv: wv_n = wv&1 (16-pos column), wv_o = wv>>1 (64 out-ch rows).
// ---------------------------------------------------------------------------
#define ROWS_ 264   // 256 bf16 + 8 pad -> 528 B row stride (16B aligned)

__global__ __launch_bounds__(256)
void fa_main(const float* __restrict__ x,
             const float* __restrict__ centerness,
             const float* __restrict__ scale,
             const float* __restrict__ offset_w,
             const float* __restrict__ offset_b,
             const float* __restrict__ mask_w,
             const float* __restrict__ mask_b,
             const float* __restrict__ conv_b,
             const unsigned short* __restrict__ wpack,
             float* __restrict__ out) {
    __shared__ int            sI[32][4];      // corner flat indices per pos
    __shared__ float          sW[32][4];      // corner weights (mask folded)
    __shared__ unsigned short samp[32][ROWS_]; // sampled bf16 [pos][c]

    const int t    = threadIdx.x;
    const int lane = t & 31;
    const int wv   = t >> 5;
    const int wv_n = wv & 1;
    const int wv_o = wv >> 1;

    const int bi    = blockIdx.x;          // 0..1023
    const int wtile = bi & 3;
    const int h     = (bi >> 2) & 127;
    const int b     = bi >> 9;
    const int w0    = wtile * 32;

    const v8f zero = {0.f, 0.f, 0.f, 0.f, 0.f, 0.f, 0.f, 0.f};
    v8f acc[4];
#pragma unroll
    for (int i = 0; i < 4; ++i) acc[i] = zero;

    const float* xb = x + (size_t)b * CHI_ * HW_;

    for (int tap = 0; tap < K_; ++tap) {
        __syncthreads();   // previous chunk's B-fragment reads done
        if (t < 32) {
            const int p = t;
            const int w = w0 + p;
            const size_t sidx = (size_t)b * HW_ + (size_t)h * W_ + w;
            const float s   = scale[sidx];
            const float cen = centerness[sidx];
            const float dy  = s * offset_w[2 * tap]     + offset_b[2 * tap];
            const float dx  = s * offset_w[2 * tap + 1] + offset_b[2 * tap + 1];
            const float m   = 1.f / (1.f + __expf(-(cen * mask_w[tap] + mask_b[tap])));
            const float ky  = (float)(tap / 3 - 1);
            const float kx  = (float)(tap % 3 - 1);
            const float ys  = (float)h + ky + dy;
            const float xs  = (float)w + kx + dx;
            const float y0f = floorf(ys);
            const float x0f = floorf(xs);
            const float wy1 = ys - y0f, wx1 = xs - x0f;
            const float wy0 = 1.f - wy1, wx0 = 1.f - wx1;
#pragma unroll
            for (int cn = 0; cn < 4; ++cn) {
                const float yc = y0f + (float)(cn >> 1);
                const float xc = x0f + (float)(cn & 1);
                const float cw = ((cn >> 1) ? wy1 : wy0) * ((cn & 1) ? wx1 : wx0);
                const bool valid = (yc >= 0.f) & (yc <= (float)(H_ - 1)) &
                                   (xc >= 0.f) & (xc <= (float)(W_ - 1));
                const int yi = (int)fminf(fmaxf(yc, 0.f), (float)(H_ - 1));
                const int xi = (int)fminf(fmaxf(xc, 0.f), (float)(W_ - 1));
                sI[p][cn] = yi * W_ + xi;
                sW[p][cn] = valid ? (cw * m) : 0.f;
            }
        }
        __syncthreads();

        // --- cooperative bilinear sampling into LDS bf16 tile [pos][c] ---
        {
            const int p = lane;                 // lane -> position (coalesced)
            const int i00 = sI[p][0], i01 = sI[p][1], i10 = sI[p][2], i11 = sI[p][3];
            const float w00 = sW[p][0], w01 = sW[p][1], w10 = sW[p][2], w11 = sW[p][3];
#pragma unroll 4
            for (int i = 0; i < 32; ++i) {
                const int c = wv * 32 + i;      // wave-uniform channel
                const float* xc = xb + (size_t)c * HW_;
                const float v = w00 * xc[i00] + w01 * xc[i01] +
                                w10 * xc[i10] + w11 * xc[i11];
                samp[p][c] = f2bf(v);
            }
        }
        __syncthreads();

        // --- GEMM: 8 K-chunks of 32 channels for this tap ---
        const int p    = wv_n * 16 + (lane & 15);
        const int csel = (lane >> 4) ? 8 : 0;
#pragma unroll
        for (int cblk = 0; cblk < 8; ++cblk) {
            const int chunk = tap * 8 + cblk;

            // B-fragment: two ds_load_b128 per lane (K runs of 8 bf16)
            Frag32B bf_;
            const uint4* bp = (const uint4*)&samp[p][cblk * 32 + csel];
            bf_.q[0] = bp[0];       // K 0..7   (or 8..15 for upper lanes)
            bf_.q[1] = bp[2];       // K 16..23 (or 24..31), +32 bytes
#pragma unroll
            for (int i = 0; i < 4; ++i) {
                const int ot = wv_o * 4 + i;
                const uint4* ap = (const uint4*)wpack +
                                  ((size_t)(chunk * 16 + ot) * 32 + lane) * 2;
                Frag32B af;
                af.q[0] = ap[0];
                af.q[1] = ap[1];
                acc[i] = __builtin_amdgcn_wmma_f32_16x16x32_bf16(
                    false, af.v, false, bf_.v, (short)0, acc[i], false, false);
            }
        }
    }

    // --- epilogue: C/D layout VGPR r -> (M = r or r+8 by lane half, N = lane&15)
    const int pn  = wv_n * 16 + (lane & 15);
    const int w   = w0 + pn;
    const int osel = (lane < 16) ? 0 : 8;
#pragma unroll
    for (int i = 0; i < 4; ++i) {
#pragma unroll
        for (int r = 0; r < 8; ++r) {
            const int o = wv_o * 64 + i * 16 + r + osel;
            out[(((size_t)b * CHO_ + o) * H_ + h) * W_ + w] = acc[i][r] + conv_b[o];
        }
    }
}

extern "C" void kernel_launch(void* const* d_in, const int* in_sizes, int n_in,
                              void* d_out, int out_size, void* d_ws, size_t ws_size,
                              hipStream_t stream) {
    const float* x          = (const float*)d_in[0];
    const float* centerness = (const float*)d_in[1];
    const float* scale      = (const float*)d_in[2];
    const float* offset_w   = (const float*)d_in[3];
    const float* offset_b   = (const float*)d_in[4];
    const float* mask_w     = (const float*)d_in[5];
    const float* mask_b     = (const float*)d_in[6];
    const float* conv_w     = (const float*)d_in[7];
    const float* conv_b     = (const float*)d_in[8];
    float* out = (float*)d_out;

    unsigned short* wpack = (unsigned short*)d_ws;   // 589824 * 2 B = 1.15 MB

    // Pass 0: weight repack into WMMA fragment order (bf16)
    fa_pack_weights<<<2304, 256, 0, stream>>>(conv_w, wpack);

    // Main fused deformable-conv implicit GEMM
    fa_main<<<1024, 256, 0, stream>>>(x, centerness, scale, offset_w, offset_b,
                                      mask_w, mask_b, conv_b, wpack, out);
}
